// BigramLM_26740466385518
// MI455X (gfx1250) — matrix operand order repaired and checked by reading, not compiled, not run on hardware
//
#include <hip/hip_runtime.h>
#include <stdint.h>

// ---------------------------------------------------------------- types
typedef __attribute__((ext_vector_type(16))) __bf16 v16bf;
typedef __attribute__((ext_vector_type(8)))  float  v8f;

union AF {
  v16bf v;
  unsigned int u32[8];
  unsigned short u16[16];
};

#define DMODEL 1024
#define NHEAD  16
#define HEADD  64
#define SEQB   4
#define SEQT   1024
#define NTOK   (SEQB * SEQT)   // 4096
#define VOCAB  32000
#define FFDIM  4096
#define NLAYER 4

__device__ __forceinline__ unsigned short f2bf(float f) {
  union { float f; unsigned u; } c; c.f = f;
  unsigned u = c.u;
  unsigned r = u + 0x7fffu + ((u >> 16) & 1u);   // round-to-nearest-even
  return (unsigned short)(r >> 16);
}

// Interleave two rows of 8 bf16 (packed in uint4s) into K-pair dwords:
// out dword c = row0[c] | (row1[c] << 16)
__device__ __forceinline__ void interleave2(uint4 lo, uint4 hi, uint4& o0,
                                            uint4& o1) {
  o0.x = (lo.x & 0xFFFFu) | (hi.x << 16);
  o0.y = (lo.x >> 16)     | (hi.x & 0xFFFF0000u);
  o0.z = (lo.y & 0xFFFFu) | (hi.y << 16);
  o0.w = (lo.y >> 16)     | (hi.y & 0xFFFF0000u);
  o1.x = (lo.z & 0xFFFFu) | (hi.z << 16);
  o1.y = (lo.z >> 16)     | (hi.z & 0xFFFF0000u);
  o1.z = (lo.w & 0xFFFFu) | (hi.w << 16);
  o1.w = (lo.w >> 16)     | (hi.w & 0xFFFF0000u);
}

// ---------------------------------------------------------------- GEMM
// C[M,N] = A[M,K](bf16 RM) * B[K,N](bf16 RM), f32 accum.
// Block tile 128x128, 8 waves (4x2), wave tile 32x64 = 2x4 WMMA frags.
// Double-buffered LDS; B tile stored K-pair-interleaved so every B-frag
// register is one aligned ds_load_b32.
#define EP_BIAS  1
#define EP_RELU  2
#define EP_RESID 4
#define EP_OUTBF 8
#define EP_OUTF  16

template <int FLAGS>
__global__ __launch_bounds__(256) void gemm_bf16(
    const unsigned short* __restrict__ A, const unsigned short* __restrict__ Bw,
    const float* __restrict__ bias, unsigned short* __restrict__ outB,
    float* __restrict__ outF, float* __restrict__ resid, int M, int N, int K) {
  constexpr int AS  = 48;    // A LDS row stride (u16), 128 rows
  constexpr int BPS = 264;   // B LDS pair-row stride (u16): 128 cols * 2 + pad
  constexpr int ASZ = 128 * AS;
  constexpr int BSZ = 16 * BPS;
  __shared__ unsigned short As[2 * ASZ];
  __shared__ unsigned short Bs[2 * BSZ];

  const int tid  = threadIdx.x;
  const int lane = tid & 31;
  const int w    = tid >> 5;
  const int wm   = (w >> 1) * 32;   // 0,32,64,96
  const int wn   = (w & 1) * 64;    // 0,64
  const int tileM = blockIdx.y * 128;
  const int tileN = blockIdx.x * 128;

  const int aRow = tid >> 1, aCol = (tid & 1) * 16;  // A: 128x32, 16 u16/thr
  const int bp   = tid >> 4, cg = (tid & 15) * 8;    // B: 16 K-pairs x 128 cols

  const int mA = lane & 15;   // frag row (A) / frag col (B,C)
  const int kh = lane >> 4;   // K-half select

  const int nt = K >> 5;      // number of 32-wide K tiles

  v8f acc[2][4] = {};
  uint4 ra0, ra1, rb0, rb1;

  auto loadG = [&](int kt) {
    const int k0 = kt * 32;
    const unsigned short* ap = A + (size_t)(tileM + aRow) * K + k0 + aCol;
    ra0 = *(const uint4*)(ap);
    ra1 = *(const uint4*)(ap + 8);
    rb0 = *(const uint4*)(Bw + (size_t)(k0 + 2 * bp) * N + tileN + cg);
    rb1 = *(const uint4*)(Bw + (size_t)(k0 + 2 * bp + 1) * N + tileN + cg);
  };
  auto storeL = [&](int buf) {
    unsigned short* Asb = As + buf * ASZ;
    unsigned short* Bsb = Bs + buf * BSZ;
    *(uint4*)(Asb + aRow * AS + aCol)     = ra0;
    *(uint4*)(Asb + aRow * AS + aCol + 8) = ra1;
    uint4 o0, o1;
    interleave2(rb0, rb1, o0, o1);
    *(uint4*)(Bsb + bp * BPS + cg * 2)     = o0;
    *(uint4*)(Bsb + bp * BPS + cg * 2 + 8) = o1;
  };

  loadG(0);
  storeL(0);
  __syncthreads();

  for (int kt = 0; kt < nt; ++kt) {
    const int cur = kt & 1;
    if (kt + 1 < nt) loadG(kt + 1);   // prefetch next tile during compute

    const unsigned short* Asb = As + cur * ASZ;
    const unsigned short* Bsb = Bs + cur * BSZ;

    AF a[2];
#pragma unroll
    for (int i = 0; i < 2; ++i) {
      const unsigned short* ap = Asb + (wm + i * 16 + mA) * AS;
#pragma unroll
      for (int j = 0; j < 4; ++j) {
        a[i].u32[j]     = *(const unsigned*)(ap + kh * 8 + 2 * j);
        a[i].u32[4 + j] = *(const unsigned*)(ap + 16 + kh * 8 + 2 * j);
      }
    }
    AF b[4];
#pragma unroll
    for (int jf = 0; jf < 4; ++jf) {
      const unsigned short* bpp = Bsb + (wn + jf * 16 + mA) * 2;
#pragma unroll
      for (int j = 0; j < 8; ++j)
        b[jf].u32[j] = *(const unsigned*)(bpp + (kh * 4 + j + (j & 4)) * BPS);
    }
#pragma unroll
    for (int i = 0; i < 2; ++i)
#pragma unroll
      for (int jf = 0; jf < 4; ++jf)
        acc[i][jf] = __builtin_amdgcn_wmma_f32_16x16x32_bf16(
            false, a[i].v, false, b[jf].v, (short)0, acc[i][jf], false, false);

    __syncthreads();                       // all reads of `cur` complete
    if (kt + 1 < nt) {
      storeL(1 - cur);
      __syncthreads();                     // next buffer visible
    }
  }

#pragma unroll
  for (int i = 0; i < 2; ++i) {
#pragma unroll
    for (int jf = 0; jf < 4; ++jf) {
#pragma unroll
      for (int r = 0; r < 8; ++r) {
        const int row = tileM + wm + i * 16 + r + 8 * kh;
        const int col = tileN + wn + jf * 16 + mA;
        float v = acc[i][jf][r];
        if (FLAGS & EP_BIAS)  v += bias[col];
        if (FLAGS & EP_RELU)  v = fmaxf(v, 0.f);
        if (FLAGS & EP_RESID) {
          float* p = resid + (size_t)row * N + col;
          *p = *p + v;
        }
        if (FLAGS & EP_OUTBF) outB[(size_t)row * N + col] = f2bf(v);
        if (FLAGS & EP_OUTF)  outF[(size_t)row * N + col] = v;
      }
    }
  }
}

// ---------------------------------------------------------------- attention
// One block = one (b,h) pair x 64 query rows. 4 waves, wave w owns rows
// [w*16, w*16+16). Online (flash) softmax over 64-key tiles, with the
// reference quirk: tril then (score==0 -> -inf).
__global__ __launch_bounds__(128) void attn_flash(
    const unsigned short* __restrict__ Q, const unsigned short* __restrict__ Kt,
    const unsigned short* __restrict__ Vt, unsigned short* __restrict__ Y) {
  constexpr int LS = 72;   // padded row stride (u16)
  __shared__ unsigned short Qs[64 * LS];
  __shared__ unsigned short Ks[64 * LS];
  __shared__ unsigned short Vs[64 * LS];
  __shared__ unsigned short Ps[64 * LS];

  const int tid = threadIdx.x, lane = tid & 31, w = tid >> 5;
  const int bh = blockIdx.x, qt = blockIdx.y;
  const int b = bh >> 4, h = bh & 15;
  const int qRow0 = b * SEQT + qt * 64;
  const int cBase = h * HEADD;
  const int mA = lane & 15, kh = lane >> 4;

  // stage Q tile (64 x 64 bf16)
  for (int it = 0; it < 4; ++it) {
    const int c = tid + it * 128;          // chunk of 8 u16
    const int row = c >> 3, cg = (c & 7) * 8;
    *(uint4*)(Qs + row * LS + cg) =
        *(const uint4*)(Q + (size_t)(qRow0 + row) * DMODEL + cBase + cg);
  }

  float mrow[8], lrow[8];
  v8f O[4] = {};
#pragma unroll
  for (int r = 0; r < 8; ++r) { mrow[r] = -INFINITY; lrow[r] = 0.f; }

  for (int j = 0; j <= qt; ++j) {
    const int kRow0 = b * SEQT + j * 64;
    __syncthreads();
    for (int it = 0; it < 4; ++it) {
      const int c = tid + it * 128;
      const int row = c >> 3, cg = (c & 7) * 8;
      *(uint4*)(Ks + row * LS + cg) =
          *(const uint4*)(Kt + (size_t)(kRow0 + row) * DMODEL + cBase + cg);
      *(uint4*)(Vs + row * LS + cg) =
          *(const uint4*)(Vt + (size_t)(kRow0 + row) * DMODEL + cBase + cg);
    }
    __syncthreads();

    // S = Q * K^T  (K^T fragment = contiguous dword pairs along K-tile rows)
    v8f s[4] = {};
#pragma unroll
    for (int ks = 0; ks < 2; ++ks) {
      AF aq;
      const unsigned short* qp = Qs + (w * 16 + mA) * LS + ks * 32;
#pragma unroll
      for (int t2 = 0; t2 < 4; ++t2) {
        aq.u32[t2]     = *(const unsigned*)(qp + kh * 8 + 2 * t2);
        aq.u32[4 + t2] = *(const unsigned*)(qp + 16 + kh * 8 + 2 * t2);
      }
#pragma unroll
      for (int jf = 0; jf < 4; ++jf) {
        AF bk;
        const unsigned short* kp = Ks + (jf * 16 + mA) * LS + ks * 32;
#pragma unroll
        for (int t2 = 0; t2 < 8; ++t2)
          bk.u32[t2] = *(const unsigned*)(kp + kh * 8 + 2 * t2 + (t2 & 4) * 2);
        s[jf] = __builtin_amdgcn_wmma_f32_16x16x32_bf16(
            false, aq.v, false, bk.v, (short)0, s[jf], false, false);
      }
    }

    // scale + causal mask + "score==0 -> -inf" quirk
    const float scale = 0.125f;   // 1/sqrt(64)
#pragma unroll
    for (int jf = 0; jf < 4; ++jf)
#pragma unroll
      for (int r = 0; r < 8; ++r) {
        const int qi = qt * 64 + w * 16 + r + 8 * kh;
        const int kj = j * 64 + jf * 16 + mA;
        float sv = s[jf][r] * scale;
        if (kj > qi || sv == 0.f) sv = -INFINITY;
        s[jf][r] = sv;
      }

    // online softmax (row stats per-lane per-reg; reduce over 16-lane half)
#pragma unroll
    for (int r = 0; r < 8; ++r) {
      float tm = fmaxf(fmaxf(s[0][r], s[1][r]), fmaxf(s[2][r], s[3][r]));
#pragma unroll
      for (int off = 1; off < 16; off <<= 1)
        tm = fmaxf(tm, __shfl_xor(tm, off, 32));
      const float mn = fmaxf(mrow[r], tm);
      const float alpha = (mrow[r] == -INFINITY) ? 0.f : __expf(mrow[r] - mn);
      float rs = 0.f;
#pragma unroll
      for (int jf = 0; jf < 4; ++jf) {
        const float p = (s[jf][r] == -INFINITY) ? 0.f : __expf(s[jf][r] - mn);
        s[jf][r] = p;
        rs += p;
      }
#pragma unroll
      for (int off = 1; off < 16; off <<= 1)
        rs += __shfl_xor(rs, off, 32);
      lrow[r] = lrow[r] * alpha + rs;
      mrow[r] = mn;
#pragma unroll
      for (int vf = 0; vf < 4; ++vf) O[vf][r] *= alpha;
    }

    // P (C layout) -> LDS -> A layout for P*V
#pragma unroll
    for (int jf = 0; jf < 4; ++jf)
#pragma unroll
      for (int r = 0; r < 8; ++r)
        Ps[(w * 16 + r + 8 * kh) * LS + jf * 16 + mA] = f2bf(s[jf][r]);
    __syncthreads();

#pragma unroll
    for (int ks = 0; ks < 2; ++ks) {
      AF ap;
      const unsigned short* pp = Ps + (w * 16 + mA) * LS + ks * 32;
#pragma unroll
      for (int t2 = 0; t2 < 4; ++t2) {
        ap.u32[t2]     = *(const unsigned*)(pp + kh * 8 + 2 * t2);
        ap.u32[4 + t2] = *(const unsigned*)(pp + 16 + kh * 8 + 2 * t2);
      }
#pragma unroll
      for (int vf = 0; vf < 4; ++vf) {
        AF bv;
#pragma unroll
        for (int t2 = 0; t2 < 8; ++t2) {
          const int kk = ks * 32 + kh * 8 + 2 * t2 + (t2 & 4) * 2;
          bv.u16[2 * t2]     = Vs[kk * LS + vf * 16 + mA];
          bv.u16[2 * t2 + 1] = Vs[(kk + 1) * LS + vf * 16 + mA];
        }
        O[vf] = __builtin_amdgcn_wmma_f32_16x16x32_bf16(
            false, ap.v, false, bv.v, (short)0, O[vf], false, false);
      }
    }
  }

  // normalize and store y (bf16, packed [tok][h*64+vdim])
#pragma unroll
  for (int r = 0; r < 8; ++r) {
    const float inv = 1.f / lrow[r];
    const int row = qRow0 + w * 16 + r + 8 * kh;
#pragma unroll
    for (int vf = 0; vf < 4; ++vf)
      Y[(size_t)row * DMODEL + cBase + vf * 16 + mA] = f2bf(O[vf][r] * inv);
  }
}

// ---------------------------------------------------------------- LN / embed
__global__ __launch_bounds__(256) void layernorm_bf16(
    const float* __restrict__ x, const float* __restrict__ g,
    const float* __restrict__ bta, unsigned short* __restrict__ out) {
  __shared__ float red[256];
  const int row = blockIdx.x, tid = threadIdx.x;
  const float* xr = x + (size_t)row * DMODEL;
  const float4 v = *(const float4*)(xr + tid * 4);
  red[tid] = v.x + v.y + v.z + v.w;
  __syncthreads();
  for (int st = 128; st > 0; st >>= 1) {
    if (tid < st) red[tid] += red[tid + st];
    __syncthreads();
  }
  const float mu = red[0] * (1.f / DMODEL);
  __syncthreads();
  const float dx = v.x - mu, dy = v.y - mu, dz = v.z - mu, dw = v.w - mu;
  red[tid] = dx * dx + dy * dy + dz * dz + dw * dw;
  __syncthreads();
  for (int st = 128; st > 0; st >>= 1) {
    if (tid < st) red[tid] += red[tid + st];
    __syncthreads();
  }
  const float rstd = rsqrtf(red[0] * (1.f / DMODEL) + 1e-5f);
  const int c = tid * 4;
  unsigned short* o = out + (size_t)row * DMODEL + c;
  o[0] = f2bf(dx * rstd * g[c + 0] + bta[c + 0]);
  o[1] = f2bf(dy * rstd * g[c + 1] + bta[c + 1]);
  o[2] = f2bf(dz * rstd * g[c + 2] + bta[c + 2]);
  o[3] = f2bf(dw * rstd * g[c + 3] + bta[c + 3]);
}

__global__ __launch_bounds__(256) void embed_kernel(
    const int* __restrict__ xids, const float* __restrict__ tok,
    const float* __restrict__ pos, float* __restrict__ hout) {
  const int row = blockIdx.x;            // b*T + t
  const int t = row & (SEQT - 1);
  const int id = xids[row];
  const int c = threadIdx.x * 4;
  const float4 tv = *(const float4*)(tok + (size_t)id * DMODEL + c);
  const float4 pv = *(const float4*)(pos + (size_t)t * DMODEL + c);
  float4 o;
  o.x = tv.x + pv.x; o.y = tv.y + pv.y; o.z = tv.z + pv.z; o.w = tv.w + pv.w;
  *(float4*)(hout + (size_t)row * DMODEL + c) = o;
}

// ---------------------------------------------------------------- weight prep
__global__ __launch_bounds__(256) void pack_headproj(
    const float* __restrict__ Wh, unsigned short* __restrict__ out) {
  // Wh[H][D][KD] -> out[d][h*64+k]  (D x D bf16)
  const int i = blockIdx.x * 256 + threadIdx.x;   // 0 .. H*D*KD-1
  const int h = i >> 16;                          // D*KD == 65536
  const int rem = i & 65535;
  const int d = rem >> 6, k = rem & 63;
  out[(size_t)d * DMODEL + h * HEADD + k] = f2bf(Wh[i]);
}

__global__ __launch_bounds__(256) void pack_wo(
    const float* __restrict__ Wo, unsigned short* __restrict__ out) {
  // Wo[H][KD][KD] -> block-diagonal D x D bf16
  const int i = blockIdx.x * 256 + threadIdx.x;   // 0 .. D*D-1
  const int r = i >> 10, c = i & 1023;
  const int hr = r >> 6, hc = c >> 6;
  const float v =
      (hr == hc) ? Wo[(size_t)hr * HEADD * HEADD + (r & 63) * HEADD + (c & 63)]
                 : 0.f;
  out[i] = f2bf(v);
}

__global__ __launch_bounds__(256) void cvt_bf16(
    const float* __restrict__ in, unsigned short* __restrict__ out, int n) {
  const int i = blockIdx.x * 256 + threadIdx.x;
  if (i < n) out[i] = f2bf(in[i]);
}

// ---------------------------------------------------------------- launch
extern "C" void kernel_launch(void* const* d_in, const int* in_sizes, int n_in,
                              void* d_out, int out_size, void* d_ws,
                              size_t ws_size, hipStream_t stream) {
  (void)in_sizes; (void)n_in; (void)out_size; (void)ws_size;
  const int*   x    = (const int*)d_in[0];
  const float* tok  = (const float*)d_in[1];
  const float* pos  = (const float*)d_in[2];
  const float* Wq   = (const float*)d_in[3];
  const float* bq   = (const float*)d_in[4];
  const float* Wk   = (const float*)d_in[5];
  const float* bk   = (const float*)d_in[6];
  const float* Wv   = (const float*)d_in[7];
  const float* bv   = (const float*)d_in[8];
  const float* Wo   = (const float*)d_in[9];
  const float* bo   = (const float*)d_in[10];
  const float* W1   = (const float*)d_in[11];
  const float* b1   = (const float*)d_in[12];
  const float* W2   = (const float*)d_in[13];
  const float* b2   = (const float*)d_in[14];
  const float* ln1g = (const float*)d_in[15];
  const float* ln1b = (const float*)d_in[16];
  const float* ln2g = (const float*)d_in[17];
  const float* ln2b = (const float*)d_in[18];
  const float* lnfg = (const float*)d_in[19];
  const float* lnfb = (const float*)d_in[20];
  const float* Wout = (const float*)d_in[21];
  const float* bout = (const float*)d_in[22];
  float* out = (float*)d_out;

  char* ws = (char*)d_ws;
  size_t off = 0;
  auto alloc = [&](size_t bytes) -> void* {
    void* p = ws + off;
    off = (off + bytes + 255) & ~(size_t)255;
    return p;
  };
  float* hbuf           = (float*)alloc((size_t)NTOK * DMODEL * 4);
  unsigned short* xnb   = (unsigned short*)alloc((size_t)NTOK * DMODEL * 2);
  unsigned short* qb    = (unsigned short*)alloc((size_t)NTOK * DMODEL * 2);
  unsigned short* kb    = (unsigned short*)alloc((size_t)NTOK * DMODEL * 2);
  unsigned short* vb    = (unsigned short*)alloc((size_t)NTOK * DMODEL * 2);
  unsigned short* yb    = (unsigned short*)alloc((size_t)NTOK * DMODEL * 2);
  unsigned short* ffb   = (unsigned short*)alloc((size_t)NTOK * FFDIM * 2);
  unsigned short* wqb   = (unsigned short*)alloc((size_t)DMODEL * DMODEL * 2);
  unsigned short* wkb   = (unsigned short*)alloc((size_t)DMODEL * DMODEL * 2);
  unsigned short* wvb   = (unsigned short*)alloc((size_t)DMODEL * DMODEL * 2);
  unsigned short* wob   = (unsigned short*)alloc((size_t)DMODEL * DMODEL * 2);
  unsigned short* w1b   = (unsigned short*)alloc((size_t)DMODEL * FFDIM * 2);
  unsigned short* w2b   = (unsigned short*)alloc((size_t)FFDIM * DMODEL * 2);
  unsigned short* woutb = (unsigned short*)alloc((size_t)DMODEL * VOCAB * 2);

  const dim3 blk(256);
  // weight conversion / packing (re-done each call; deterministic)
  pack_headproj<<<(NHEAD * DMODEL * HEADD) / 256, blk, 0, stream>>>(Wq, wqb);
  pack_headproj<<<(NHEAD * DMODEL * HEADD) / 256, blk, 0, stream>>>(Wk, wkb);
  pack_headproj<<<(NHEAD * DMODEL * HEADD) / 256, blk, 0, stream>>>(Wv, wvb);
  pack_wo<<<(DMODEL * DMODEL) / 256, blk, 0, stream>>>(Wo, wob);
  cvt_bf16<<<(DMODEL * FFDIM) / 256, blk, 0, stream>>>(W1, w1b, DMODEL * FFDIM);
  cvt_bf16<<<(FFDIM * DMODEL) / 256, blk, 0, stream>>>(W2, w2b, FFDIM * DMODEL);
  cvt_bf16<<<(DMODEL * VOCAB) / 256, blk, 0, stream>>>(Wout, woutb,
                                                       DMODEL * VOCAB);

  embed_kernel<<<NTOK, blk, 0, stream>>>(x, tok, pos, hbuf);

  const dim3 gq(DMODEL / 128, NTOK / 128);
  for (int layer = 0; layer < NLAYER; ++layer) {
    layernorm_bf16<<<NTOK, blk, 0, stream>>>(hbuf, ln1g, ln1b, xnb);
    gemm_bf16<EP_BIAS | EP_OUTBF><<<gq, blk, 0, stream>>>(
        xnb, wqb, bq, qb, nullptr, nullptr, NTOK, DMODEL, DMODEL);
    gemm_bf16<EP_BIAS | EP_OUTBF><<<gq, blk, 0, stream>>>(
        xnb, wkb, bk, kb, nullptr, nullptr, NTOK, DMODEL, DMODEL);
    gemm_bf16<EP_BIAS | EP_OUTBF><<<gq, blk, 0, stream>>>(
        xnb, wvb, bv, vb, nullptr, nullptr, NTOK, DMODEL, DMODEL);
    attn_flash<<<dim3(SEQB * NHEAD, SEQT / 64), dim3(128), 0, stream>>>(
        qb, kb, vb, yb);
    gemm_bf16<EP_BIAS | EP_RESID><<<gq, blk, 0, stream>>>(
        yb, wob, bo, nullptr, nullptr, hbuf, NTOK, DMODEL, DMODEL);
    layernorm_bf16<<<NTOK, blk, 0, stream>>>(hbuf, ln2g, ln2b, xnb);
    gemm_bf16<EP_BIAS | EP_RELU | EP_OUTBF>
        <<<dim3(FFDIM / 128, NTOK / 128), blk, 0, stream>>>(
            xnb, w1b, b1, ffb, nullptr, nullptr, NTOK, FFDIM, DMODEL);
    gemm_bf16<EP_BIAS | EP_RESID><<<gq, blk, 0, stream>>>(
        ffb, w2b, b2, nullptr, nullptr, hbuf, NTOK, DMODEL, FFDIM);
  }
  layernorm_bf16<<<NTOK, blk, 0, stream>>>(hbuf, lnfg, lnfb, xnb);
  gemm_bf16<EP_BIAS | EP_OUTF>
      <<<dim3(VOCAB / 128, NTOK / 128), blk, 0, stream>>>(
          xnb, woutb, bout, nullptr, out, nullptr, NTOK, VOCAB, DMODEL);
}
